// MultiheadAttention_12206297055886
// MI455X (gfx1250) — matrix operand order repaired and checked by reading, not compiled, not run on hardware
//
#include <hip/hip_runtime.h>
#include <hip/hip_bf16.h>

typedef __attribute__((ext_vector_type(16))) _Float16 v16h;
typedef __attribute__((ext_vector_type(8)))  _Float16 v8h;
typedef __attribute__((ext_vector_type(8)))  float    v8f;
typedef __attribute__((ext_vector_type(4)))  float    v4f;

#define BB 2
#define NN 2048
#define DD 128
#define HH 16
#define AA 2048   // DD*HH

// ---------------- WMMA helpers (CDNA5 16x16x32 f16 -> f32) ----------------

__device__ __forceinline__ v8f wmma_f16(v16h a, v16h b, v8f c) {
  // (neg_a, A, neg_b, B, c_mod, C, reuse_a, reuse_b)
  return __builtin_amdgcn_wmma_f32_16x16x32_f16(false, a, false, b, (short)0, c,
                                                false, false);
}

// A-fragment (16x32) from row-major [row][k] f16, row stride ld.
// Lane L: row = row0 + (L&15); halves 0..7 = K kb..kb+7, 8..15 = K kb+16..kb+23,
// where kb = kbase + 8*(L>=16).  Matches ISA 16-bit A layout table.
__device__ __forceinline__ v16h load_a16(const _Float16* base, int ld, int row0,
                                         int kbase, int lane) {
  const _Float16* p = base + (size_t)(row0 + (lane & 15)) * ld + kbase + ((lane >> 4) << 3);
  v8h lo = *(const v8h*)p;
  v8h hi = *(const v8h*)(p + 16);
  v16h r;
#pragma unroll
  for (int i = 0; i < 8; ++i) { r[i] = lo[i]; r[i + 8] = hi[i]; }
  return r;
}

// B-fragment (32x16) from [n][k] storage (n = output column), row stride ld.
// Lane L: n = n0 + (L&15); halves 0..7 = K kb..kb+7, 8..15 = K kb+16..kb+23, kb = kbase + 8*(L>=16).
__device__ __forceinline__ v16h load_b16(const _Float16* base, int ld, int n0,
                                         int kbase, int lane) {
  const _Float16* p = base + (size_t)(n0 + (lane & 15)) * ld + kbase + ((lane >> 4) << 3);
  v8h lo = *(const v8h*)p;
  v8h hi = *(const v8h*)(p + 16);
  v16h r;
#pragma unroll
  for (int i = 0; i < 8; ++i) { r[i] = lo[i]; r[i + 8] = hi[i]; }
  return r;
}

// ---------------- kernel 1: transpose/convert weights to f16 ----------------
// z=0..2: W[128][2048] -> WT[2048][128];  z=3: Wo[2048][128] -> WoT[128][2048]
__global__ void transpose_w(const float* __restrict__ Wq, const float* __restrict__ Wk,
                            const float* __restrict__ Wv, const float* __restrict__ Wo,
                            _Float16* __restrict__ WTq, _Float16* __restrict__ WTk,
                            _Float16* __restrict__ WTv, _Float16* __restrict__ WoT) {
  const int z = blockIdx.z;
  const int idx = blockIdx.x * 256 + threadIdx.x;   // 0..262143
  if (z < 3) {
    const float* W  = (z == 0) ? Wq : (z == 1) ? Wk : Wv;
    _Float16*   WT  = (z == 0) ? WTq : (z == 1) ? WTk : WTv;
    const int c = idx >> 7, k = idx & 127;          // WT[c][k]
    WT[idx] = (_Float16)W[(size_t)k * AA + c];
  } else {
    const int c = idx >> 11, k = idx & 2047;        // WoT[c][k]
    WoT[idx] = (_Float16)Wo[(size_t)k * DD + c];
  }
}

// ---------------- kernel 2: QKV projection GEMM -> head-major f16 -----------
__global__ void __launch_bounds__(128)
proj_qkv(const float* __restrict__ x,
         const _Float16* __restrict__ WTq, const _Float16* __restrict__ WTk,
         const _Float16* __restrict__ WTv,
         const float* __restrict__ bq, const float* __restrict__ bk,
         const float* __restrict__ bv,
         _Float16* __restrict__ Qo, _Float16* __restrict__ Ko,
         _Float16* __restrict__ Vo) {
  const int z = blockIdx.z;
  const _Float16* WT = (z == 0) ? WTq : (z == 1) ? WTk : WTv;
  const float* bias  = (z == 0) ? bq : (z == 1) ? bk : bv;
  _Float16* dst      = (z == 0) ? Qo : (z == 1) ? Ko : Vo;

  const int wv = threadIdx.x >> 5, lane = threadIdx.x & 31;
  const int hl = lane >> 4, ln = lane & 15;
  const int m0 = blockIdx.x * 16;                 // row tile over B*N = 4096
  const int n0 = (blockIdx.y * 4 + wv) * 16;      // col tile over ALL = 2048

  v8f acc = {0.f, 0.f, 0.f, 0.f, 0.f, 0.f, 0.f, 0.f};
#pragma unroll
  for (int c = 0; c < 4; ++c) {
    // A fragment from fp32 x, converted to f16 on the fly
    const float* xr = x + (size_t)(m0 + ln) * DD + c * 32 + (hl << 3);
    v4f f0 = *(const v4f*)(xr);
    v4f f1 = *(const v4f*)(xr + 4);
    v4f f2 = *(const v4f*)(xr + 16);
    v4f f3 = *(const v4f*)(xr + 20);
    v16h a;
#pragma unroll
    for (int i = 0; i < 4; ++i) {
      a[i]      = (_Float16)f0[i];
      a[4 + i]  = (_Float16)f1[i];
      a[8 + i]  = (_Float16)f2[i];
      a[12 + i] = (_Float16)f3[i];
    }
    v16h bfrag = load_b16(WT, DD, n0, c * 32, lane);
    acc = wmma_f16(a, bfrag, acc);
  }
  const float bcol = bias[n0 + ln];
#pragma unroll
  for (int r = 0; r < 8; ++r) {
    const int m  = m0 + r + (hl << 3);
    const int bt = m >> 11, n = m & (NN - 1);
    const int ct = n0 + ln;
    const int h  = ct >> 7, dd = ct & (DD - 1);
    dst[(((size_t)(bt * HH + h)) * NN + n) * DD + dd] = (_Float16)(acc[r] + bcol);
  }
}

// ---------------- kernel 3: flash attention ---------------------------------
// grid = B*H*(N/64) blocks, 4 waves/block, one wave = 16 query rows.
// Software-pipelined: next K/V tile staged in registers while current tile
// feeds the WMMAs; tile after next pre-warmed via global_prefetch.
__global__ void __launch_bounds__(128)
attn_kernel(const _Float16* __restrict__ Qf, const _Float16* __restrict__ Kf,
            const _Float16* __restrict__ Vf, const int* __restrict__ mask,
            _Float16* __restrict__ Of) {
  __shared__ __align__(16) _Float16 sK[32 * DD];    // K tile, [key][dim]
  __shared__ __align__(16) _Float16 sVt[DD * 32];   // V tile transposed, [dim][key]
  __shared__ __align__(16) _Float16 sP[4 * 16 * 32];// per-wave P scratch

  const int tid = threadIdx.x;
  const int wv = tid >> 5, lane = tid & 31;
  const int hl = lane >> 4, ln = lane & 15;

  const int bid  = blockIdx.x;
  const int b    = bid / (HH * (NN / 64));
  const int rem  = bid % (HH * (NN / 64));
  const int h    = rem / (NN / 64);
  const int qblk = rem % (NN / 64);
  const int q0   = qblk * 64 + wv * 16;

  const size_t headOff = ((size_t)(b * HH + h)) * NN * DD;
  const _Float16* Qh = Qf + headOff;
  const _Float16* Kh = Kf + headOff;
  const _Float16* Vh = Vf + headOff;
  const int* mrow = mask + (size_t)b * NN;

  v16h aq[4];
#pragma unroll
  for (int c = 0; c < 4; ++c) aq[c] = load_a16(Qh, DD, q0, c * 32, lane);

  v8f acc[8];
  float mprev[8], lsum[8];
#pragma unroll
  for (int c = 0; c < 8; ++c)
#pragma unroll
    for (int r = 0; r < 8; ++r) acc[c][r] = 0.f;
#pragma unroll
  for (int r = 0; r < 8; ++r) { mprev[r] = -1e30f; lsum[r] = 0.f; }

  const float scale = 0.08838834764831845f;  // 1/sqrt(128)

  // register staging for next K/V tile: 4 x 16B per matrix per thread
  v8h kreg[4], vreg[4];
  {
    const _Float16* Ksrc = Kh;
    const _Float16* Vsrc = Vh;
#pragma unroll
    for (int i = 0; i < 4; ++i) {
      kreg[i] = *(const v8h*)(Ksrc + (size_t)(tid + i * 128) * 8);
      vreg[i] = *(const v8h*)(Vsrc + (size_t)(tid + i * 128) * 8);
    }
  }

  for (int jb = 0; jb < NN; jb += 32) {
    __syncthreads();  // previous iteration done reading sK/sVt
    // publish staged tile to LDS (K row-major, V transposed)
#pragma unroll
    for (int i = 0; i < 4; ++i) {
      const int idx = (tid + i * 128) * 8;       // first half index of chunk
      *(v8h*)(sK + idx) = kreg[i];
      const int key = idx >> 7, dim0 = idx & (DD - 1);
#pragma unroll
      for (int j = 0; j < 8; ++j) sVt[(dim0 + j) * 32 + key] = vreg[i][j];
    }
    __syncthreads();

    // stage next tile (overlaps with compute below); uniform branch
    if (jb + 32 < NN) {
      const _Float16* Ksrc = Kh + (size_t)(jb + 32) * DD;
      const _Float16* Vsrc = Vh + (size_t)(jb + 32) * DD;
#pragma unroll
      for (int i = 0; i < 4; ++i) {
        kreg[i] = *(const v8h*)(Ksrc + (size_t)(tid + i * 128) * 8);
        vreg[i] = *(const v8h*)(Vsrc + (size_t)(tid + i * 128) * 8);
      }
    }
    // pre-warm tile after next (global_prefetch path)
    if (jb + 64 < NN) {
      __builtin_prefetch(Kh + (size_t)(jb + 64) * DD + tid * 32, 0, 0);
      __builtin_prefetch(Vh + (size_t)(jb + 64) * DD + tid * 32, 0, 0);
    }

    // S = Q * K^T  (two 16x16 tiles: keys 0..15 and 16..31 of the block)
    v8f s0 = {0.f, 0.f, 0.f, 0.f, 0.f, 0.f, 0.f, 0.f};
    v8f s1 = {0.f, 0.f, 0.f, 0.f, 0.f, 0.f, 0.f, 0.f};
#pragma unroll
    for (int c = 0; c < 4; ++c) {
      v16h b0 = load_b16(sK, DD, 0, c * 32, lane);
      s0 = wmma_f16(aq[c], b0, s0);
    }
#pragma unroll
    for (int c = 0; c < 4; ++c) {
      v16h b1 = load_b16(sK, DD, 16, c * 32, lane);
      s1 = wmma_f16(aq[c], b1, s1);
    }

    const int km0 = mrow[jb + ln];
    const int km1 = mrow[jb + 16 + ln];

    _Float16* sPw = sP + wv * (16 * 32);
#pragma unroll
    for (int r = 0; r < 8; ++r) {
      float f0 = km0 ? s0[r] * scale : -1e30f;
      float f1 = km1 ? s1[r] * scale : -1e30f;
      // row max across the 16 lanes holding this row (xor<=8 stays in half)
      float t = fmaxf(f0, f1);
      t = fmaxf(t, __shfl_xor(t, 1, 32));
      t = fmaxf(t, __shfl_xor(t, 2, 32));
      t = fmaxf(t, __shfl_xor(t, 4, 32));
      t = fmaxf(t, __shfl_xor(t, 8, 32));
      const float mnew = fmaxf(mprev[r], t);
      const float p0 = __expf(f0 - mnew);
      const float p1 = __expf(f1 - mnew);
      float rs = p0 + p1;
      rs += __shfl_xor(rs, 1, 32);
      rs += __shfl_xor(rs, 2, 32);
      rs += __shfl_xor(rs, 4, 32);
      rs += __shfl_xor(rs, 8, 32);
      const float alpha = __expf(mprev[r] - mnew);
      lsum[r] = lsum[r] * alpha + rs;
      mprev[r] = mnew;
#pragma unroll
      for (int c = 0; c < 8; ++c) acc[c][r] *= alpha;
      const int row = r + (hl << 3);
      sPw[row * 32 + ln]      = (_Float16)p0;
      sPw[row * 32 + 16 + ln] = (_Float16)p1;
    }
    __syncthreads();  // make P visible cross-lane via LDS

    // acc += P (16x32) * V (32x128)
    v16h ap = load_a16(sPw, 32, 0, 0, lane);
#pragma unroll
    for (int c = 0; c < 8; ++c) {
      v16h bvf = load_b16(sVt, 32, c * 16, 0, lane);
      acc[c] = wmma_f16(ap, bvf, acc[c]);
    }
  }

  // finalize: 1/l, query mask, store f16 to [b*N+n][h*128+dd]
#pragma unroll
  for (int r = 0; r < 8; ++r) {
    const int qrow = q0 + r + (hl << 3);
    const int qm = mrow[qrow];
    const float inv = (qm != 0 && lsum[r] > 0.f) ? (1.f / lsum[r]) : 0.f;
    const size_t obase = ((size_t)(b * NN + qrow)) * AA + h * DD;
#pragma unroll
    for (int c = 0; c < 8; ++c)
      Of[obase + c * 16 + ln] = (_Float16)(acc[c][r] * inv);
  }
}

// ---------------- kernel 4: output projection -------------------------------
__global__ void __launch_bounds__(128)
out_proj(const _Float16* __restrict__ O, const _Float16* __restrict__ WoT,
         const float* __restrict__ bo, float* __restrict__ out) {
  const int wv = threadIdx.x >> 5, lane = threadIdx.x & 31;
  const int hl = lane >> 4, ln = lane & 15;
  const int tile = blockIdx.x * 4 + wv;   // 2048 tiles (256 M x 8 N)
  const int m0 = (tile >> 3) * 16;
  const int n0 = (tile & 7) * 16;
  v8f acc = {0.f, 0.f, 0.f, 0.f, 0.f, 0.f, 0.f, 0.f};
  for (int kb = 0; kb < AA; kb += 32) {
    v16h a  = load_a16(O, AA, m0, kb, lane);
    v16h bf = load_b16(WoT, AA, n0, kb, lane);
    acc = wmma_f16(a, bf, acc);
  }
  const float bcol = bo[n0 + ln];
#pragma unroll
  for (int r = 0; r < 8; ++r)
    out[(size_t)(m0 + r + (hl << 3)) * DD + n0 + ln] = acc[r] + bcol;
}

// ---------------- launch ----------------------------------------------------
extern "C" void kernel_launch(void* const* d_in, const int* in_sizes, int n_in,
                              void* d_out, int out_size, void* d_ws, size_t ws_size,
                              hipStream_t stream) {
  (void)in_sizes; (void)n_in; (void)out_size; (void)ws_size;
  const float* x  = (const float*)d_in[0];
  const int* mask = (const int*)d_in[1];
  const float* Wq = (const float*)d_in[2];
  const float* bq = (const float*)d_in[3];
  const float* Wk = (const float*)d_in[4];
  const float* bk = (const float*)d_in[5];
  const float* Wv = (const float*)d_in[6];
  const float* bv = (const float*)d_in[7];
  const float* Wo = (const float*)d_in[8];
  const float* bo = (const float*)d_in[9];
  float* out = (float*)d_out;

  _Float16* ws = (_Float16*)d_ws;
  const size_t nBig = (size_t)BB * HH * NN * DD;  // 8,388,608 halves
  _Float16* Qf  = ws;
  _Float16* Kf  = Qf + nBig;
  _Float16* Vf  = Kf + nBig;
  _Float16* Of  = Vf + nBig;
  _Float16* WTq = Of + nBig;
  _Float16* WTk = WTq + (size_t)AA * DD;
  _Float16* WTv = WTk + (size_t)AA * DD;
  _Float16* WoT = WTv + (size_t)AA * DD;

  transpose_w<<<dim3(1024, 1, 4), 256, 0, stream>>>(Wq, Wk, Wv, Wo, WTq, WTk, WTv, WoT);
  proj_qkv<<<dim3(256, 32, 3), 128, 0, stream>>>(x, WTq, WTk, WTv, bq, bk, bv, Qf, Kf, Vf);
  attn_kernel<<<dim3(BB * HH * (NN / 64)), 128, 0, stream>>>(Qf, Kf, Vf, mask, Of);
  out_proj<<<dim3(512), 128, 0, stream>>>(Of, WoT, bo, out);
}
